// FSQCodebook_72395968741706
// MI455X (gfx1250) — matrix operand order repaired and checked by reading, not compiled, not run on hardware
//
#include <hip/hip_runtime.h>
#include <hip/hip_bf16.h>

typedef __attribute__((ext_vector_type(16))) _Float16 v16h;
typedef __attribute__((ext_vector_type(8)))  _Float16 v8h;
typedef __attribute__((ext_vector_type(8)))  float    v8f;

#define DDIM    1280
#define KCHUNK  32
#define LDSW    1288   /* padded row stride in halves: 644 dwords/row -> 4-bank skew, conflict-free b128 */
#define SCALE_F 0.9990000128746033f

__global__ __launch_bounds__(256) void fsq_wmma_kernel(const float* __restrict__ x,
                                                       const float* __restrict__ W,
                                                       const float* __restrict__ bias_p,
                                                       int* __restrict__ out)
{
    // W staged as f16, 16 rows (8 real + 8 zero) so B's padded columns are exact zeros.
    __shared__ _Float16 Wl[16 * LDSW];

    const int tid = threadIdx.x;
    for (int i = tid; i < 16 * DDIM; i += 256) {
        int n = i / DDIM;
        int d = i - n * DDIM;
        Wl[n * LDSW + d] = (n < 8) ? (_Float16)W[n * DDIM + d] : (_Float16)0.0f;
    }
    __syncthreads();

    const int lane  = tid & 31;
    const int wave  = tid >> 5;
    const int row0  = (blockIdx.x * 8 + wave) * 16;  // 16 rows of x per wave
    const int m     = lane & 15;                      // A row / B col / D col index
    const int hi    = lane >> 4;                      // half-wave select
    const int abase = hi * 8;                         // A-layout K sub-offset
    const int bbase = hi * 16;                        // unused; B loads use abase like A

    const float* xp   = x + (size_t)(row0 + m) * DDIM + abase;
    const int    woff = m * LDSW;

    v8f c = {};

    #pragma unroll 4
    for (int k0 = 0; k0 < DDIM; k0 += KCHUNK) {
        const float* xq = xp + k0;
        __builtin_prefetch(xq + 32 * KCHUNK, 0, 0);   // stream ~4KB ahead

        // A fragment: lanes<16 hold K = k0+{0..7, 16..23}; lanes>=16 hold +8 (ISA 16-bit A layout)
        float4 f0 = *(const float4*)(xq);
        float4 f1 = *(const float4*)(xq + 4);
        float4 f2 = *(const float4*)(xq + 16);
        float4 f3 = *(const float4*)(xq + 20);

        v16h a;
        a[0]  = (_Float16)f0.x; a[1]  = (_Float16)f0.y; a[2]  = (_Float16)f0.z; a[3]  = (_Float16)f0.w;
        a[4]  = (_Float16)f1.x; a[5]  = (_Float16)f1.y; a[6]  = (_Float16)f1.z; a[7]  = (_Float16)f1.w;
        a[8]  = (_Float16)f2.x; a[9]  = (_Float16)f2.y; a[10] = (_Float16)f2.z; a[11] = (_Float16)f2.w;
        a[12] = (_Float16)f3.x; a[13] = (_Float16)f3.y; a[14] = (_Float16)f3.z; a[15] = (_Float16)f3.w;

        // B fragment: column-major; same K order as A (lanes 0-15 -> K=k0+{0..7,16..23},
        // lanes 16-31 -> K=k0+{8..15,24..31})
        v8h blo = *(const v8h*)&Wl[woff + k0 + abase];
        v8h bhi = *(const v8h*)&Wl[woff + k0 + abase + 16];
        v16h bm = __builtin_shufflevector(blo, bhi,
                    0, 1, 2, 3, 4, 5, 6, 7, 8, 9, 10, 11, 12, 13, 14, 15);

        c = __builtin_amdgcn_wmma_f32_16x16x32_f16(false, a, false, bm,
                                                   (short)0, c, false, false);
    }

    // Epilogue: lane's column N = m; pow3 = 3^N for N<8 else 0 (padded cols contribute nothing).
    float bv = (m < 8) ? bias_p[m] : 0.0f;
    float p3 = 0.0f;
    if (m < 8) { p3 = 1.0f; for (int i = 0; i < m; ++i) p3 *= 3.0f; }

    float sums[8];
    #pragma unroll
    for (int r = 0; r < 8; ++r) {
        float h = c[r] + bv;
        float t = tanhf(h) * SCALE_F;
        float v = (rintf(t) + 1.0f) * p3;     // ternary {0,1,2} * 3^N
        v += __shfl_xor(v, 1, 32);            // reduce over 8-lane group
        v += __shfl_xor(v, 2, 32);
        v += __shfl_xor(v, 4, 32);
        sums[r] = v;                          // lane 0: row r; lane 16: row r+8
    }

    if (m == 0) {
        int obase = row0 + 8 * hi;
        int4 o0 = make_int4((int)sums[0], (int)sums[1], (int)sums[2], (int)sums[3]);
        int4 o1 = make_int4((int)sums[4], (int)sums[5], (int)sums[6], (int)sums[7]);
        *(int4*)(out + obase)     = o0;
        *(int4*)(out + obase + 4) = o1;
    }
}

extern "C" void kernel_launch(void* const* d_in, const int* in_sizes, int n_in,
                              void* d_out, int out_size, void* d_ws, size_t ws_size,
                              hipStream_t stream) {
    (void)n_in; (void)d_ws; (void)ws_size; (void)out_size;
    const float* x = (const float*)d_in[0];   // (8, 8192, 1280) fp32
    const float* W = (const float*)d_in[1];   // (8, 1280) fp32
    const float* b = (const float*)d_in[2];   // (8,) fp32
    int* out = (int*)d_out;                   // (8, 8192) int32

    const int rows   = in_sizes[0] / DDIM;    // 65536
    const int blocks = rows / 128;            // 512 blocks * 8 waves * 16 rows
    fsq_wmma_kernel<<<blocks, 256, 0, stream>>>(x, W, b, out);
}
